// HierarchicalAttentionPooling_33861522162213
// MI455X (gfx1250) — matrix operand order repaired and checked by reading, not compiled, not run on hardware
//
#include <hip/hip_runtime.h>
#include <math.h>

#define NSEG    1024
#define NHEADS  4
#define HID     64
#define DIM     512
#define NNODES  262144
#define HK      256   /* NHEADS*HID */

typedef __attribute__((ext_vector_type(16))) __bf16 v16bf;
typedef __attribute__((ext_vector_type(8)))  __bf16 v8bf;
typedef __attribute__((ext_vector_type(8)))  float  v8f;

// Workspace layout (bytes)
#define WH_OFF  ((size_t)0)                         // 131072 bf16 = 256 KB
#define WC_OFF  ((size_t)262144)                    // 1048576 bf16 = 2 MB
#define SEG_OFF ((size_t)2359296)                   // 1025 int
#define SC_OFF  ((size_t)2363648)                   // N*4 f32 = 4 MB (scores -> weights)
#define PL_OFF  ((size_t)6557952)                   // 1024*2048 bf16 = 4 MB

__device__ __forceinline__ v8f wmma_bf16(v16bf a, v16bf b, v8f c) {
#if defined(__gfx1250__)
  return __builtin_amdgcn_wmma_f32_16x16x32_bf16(false, a, false, b, (short)0, c,
                                                 false, false);
#else
  return c;  // host-pass placeholder, never executed
#endif
}

// Async cache->LDS copy (16B per lane), tracked by ASYNCcnt (CDNA5).
__device__ __forceinline__ void async_copy_b128(unsigned int lds_byte_addr,
                                                const void* gaddr) {
#if defined(__gfx1250__)
  asm volatile("global_load_async_to_lds_b128 %0, %1, off"
               :
               : "v"(lds_byte_addr), "v"((unsigned long long)(uintptr_t)gaddr)
               : "memory");
#else
  (void)lds_byte_addr; (void)gaddr;
#endif
}

__device__ __forceinline__ void wait_async0() {
#if defined(__gfx1250__)
  asm volatile("s_wait_asynccnt 0x0" ::: "memory");
#endif
}

// ---------------------------------------------------------------- convert
__global__ void cvt_f32_bf16_kernel(const float* __restrict__ src,
                                    __bf16* __restrict__ dst, int n) {
  int i = blockIdx.x * blockDim.x + threadIdx.x;
  if (i < n) dst[i] = (__bf16)src[i];
}

// ---------------------------------------------------------------- segment bounds
__global__ void seg_bounds_kernel(const long long* __restrict__ batch,
                                  int* __restrict__ seg) {
  const int s = blockIdx.x * blockDim.x + threadIdx.x;
  if (s > NSEG) return;
  int lo = 0, hi = NNODES;
  while (lo < hi) {
    int mid = (lo + hi) >> 1;
    if (batch[mid] < (long long)s) lo = mid + 1; else hi = mid;
  }
  seg[s] = lo;  // lower bound; seg[NSEG] == NNODES
}

// ---------------------------------------------------------------- scores (WMMA)
// Each wave: 16 nodes. logits[16,256] = x_tile[16,512] * Wh^T, then
// score[n,h] = sum_k tanh(logit + bh) * ctx.
__global__ __launch_bounds__(256)
void scores_kernel(const float* __restrict__ x,
                   const __bf16* __restrict__ whb,   // [256,512] bf16 row-major
                   const float* __restrict__ bh,     // [256]
                   const float* __restrict__ ctx,    // [256]
                   float* __restrict__ scores) {     // [N,4]
  __shared__ __align__(64) __bf16 smemB[HK * 32];    // 16 KB: 256 cols x 32 k
  const int tid  = threadIdx.x;
  const int lane = tid & 31;
  const int wave = tid >> 5;
  const int tileBase = blockIdx.x * 128 + wave * 16;
  const int mrow = tileBase + (lane & 15);
  const float* xrow = x + (size_t)mrow * DIM;
  const int kb    = (lane < 16) ? 0 : 16;   // A-fragment K half per lane group (contiguous k = 16h + s)
  const int bbase = (lane < 16) ? 0 : 16;   // B-fragment K half per lane group

  const v8f vz = {0.f, 0.f, 0.f, 0.f, 0.f, 0.f, 0.f, 0.f};
  v8f acc[16];
#pragma unroll
  for (int c = 0; c < 16; ++c) acc[c] = vz;

  for (int ks = 0; ks < 16; ++ks) {
    __syncthreads();  // waits DScnt: prior iteration's B-fragment reads done
    {  // stage Wh slab: 256 cols x 32 K bf16 -> LDS via async cache->LDS path
      const __bf16* gsrc = whb + (size_t)tid * DIM + ks * 32;
      const unsigned int ldst =
          (unsigned int)(uintptr_t)(const void*)(smemB + tid * 32);
      async_copy_b128(ldst,      gsrc);
      async_copy_b128(ldst + 16, gsrc + 8);
      async_copy_b128(ldst + 32, gsrc + 16);
      async_copy_b128(ldst + 48, gsrc + 24);
    }

    const int k0 = ks * 32 + kb;
    __builtin_prefetch(xrow + k0 + 64, 0, 1);  // speculative: two slabs ahead
    // A fragment: this code's contiguous K order (as B) -> lane holds K = k0..k0+15
    float4 f0 = *(const float4*)(xrow + k0);
    float4 f1 = *(const float4*)(xrow + k0 + 4);
    float4 f2 = *(const float4*)(xrow + k0 + 8);
    float4 f3 = *(const float4*)(xrow + k0 + 12);
    v16bf a;
    a[0]=(__bf16)f0.x;  a[1]=(__bf16)f0.y;  a[2]=(__bf16)f0.z;  a[3]=(__bf16)f0.w;
    a[4]=(__bf16)f1.x;  a[5]=(__bf16)f1.y;  a[6]=(__bf16)f1.z;  a[7]=(__bf16)f1.w;
    a[8]=(__bf16)f2.x;  a[9]=(__bf16)f2.y;  a[10]=(__bf16)f2.z; a[11]=(__bf16)f2.w;
    a[12]=(__bf16)f3.x; a[13]=(__bf16)f3.y; a[14]=(__bf16)f3.z; a[15]=(__bf16)f3.w;

    wait_async0();    // own wave's async transfers landed in LDS
    __syncthreads();  // all waves' slab portions visible

#pragma unroll
    for (int c = 0; c < 16; ++c) {
      const v16bf b =
          *(const v16bf*)(smemB + (c * 16 + (lane & 15)) * 32 + bbase);
      acc[c] = wmma_bf16(a, b, acc[c]);
    }
  }

  // Epilogue: sc[h][r] = sum over this lane's columns of tanh(logit+bh)*ctx
  float sc[NHEADS][8];
#pragma unroll
  for (int h = 0; h < NHEADS; ++h)
#pragma unroll
    for (int r = 0; r < 8; ++r) sc[h][r] = 0.0f;

#pragma unroll
  for (int c = 0; c < 16; ++c) {
    const int col = c * 16 + (lane & 15);
    const float cv = ctx[col];
    const float bv = bh[col];
    const int h = c >> 2;  // 4 column-tiles per head
#pragma unroll
    for (int r = 0; r < 8; ++r)
      sc[h][r] += tanhf(acc[c][r] + bv) * cv;
  }

  // reduce across the 16 lanes sharing a row group (C/D layout: halves)
#pragma unroll
  for (int h = 0; h < NHEADS; ++h)
#pragma unroll
    for (int r = 0; r < 8; ++r) {
      float v = sc[h][r];
      v += __shfl_xor(v, 1, 32);
      v += __shfl_xor(v, 2, 32);
      v += __shfl_xor(v, 4, 32);
      v += __shfl_xor(v, 8, 32);
      sc[h][r] = v;
    }

  if ((lane & 15) == 0) {
    const int rbase = (lane < 16) ? 0 : 8;  // lanes 0-15: rows 0-7; 16-31: 8-15
#pragma unroll
    for (int r = 0; r < 8; ++r) {
      const int node = tileBase + rbase + r;
#pragma unroll
      for (int h = 0; h < NHEADS; ++h)
        scores[(size_t)node * NHEADS + h] = sc[h][r];
    }
  }
}

// ---------------------------------------------------------------- segment softmax
__global__ __launch_bounds__(256)
void softmax_kernel(float* __restrict__ scores, const int* __restrict__ seg) {
  const int s = blockIdx.x;
  const int s0 = seg[s], s1 = seg[s + 1];
  const int tid = threadIdx.x;
  __shared__ float red[256 * NHEADS];

  // phase 1: per-head max
  float mx[NHEADS];
#pragma unroll
  for (int h = 0; h < NHEADS; ++h) mx[h] = -3.402823e38f;
  for (int i = s0 + tid; i < s1; i += 256)
#pragma unroll
    for (int h = 0; h < NHEADS; ++h)
      mx[h] = fmaxf(mx[h], scores[(size_t)i * NHEADS + h]);
#pragma unroll
  for (int h = 0; h < NHEADS; ++h) red[tid * NHEADS + h] = mx[h];
  __syncthreads();
  for (int off = 128; off > 0; off >>= 1) {
    if (tid < off)
#pragma unroll
      for (int h = 0; h < NHEADS; ++h)
        red[tid * NHEADS + h] =
            fmaxf(red[tid * NHEADS + h], red[(tid + off) * NHEADS + h]);
    __syncthreads();
  }
  float smax[NHEADS];
#pragma unroll
  for (int h = 0; h < NHEADS; ++h) smax[h] = red[h];
  __syncthreads();

  // phase 2: sum of exp
  float sm[NHEADS];
#pragma unroll
  for (int h = 0; h < NHEADS; ++h) sm[h] = 0.0f;
  for (int i = s0 + tid; i < s1; i += 256)
#pragma unroll
    for (int h = 0; h < NHEADS; ++h)
      sm[h] += __expf(scores[(size_t)i * NHEADS + h] - smax[h]);
#pragma unroll
  for (int h = 0; h < NHEADS; ++h) red[tid * NHEADS + h] = sm[h];
  __syncthreads();
  for (int off = 128; off > 0; off >>= 1) {
    if (tid < off)
#pragma unroll
      for (int h = 0; h < NHEADS; ++h)
        red[tid * NHEADS + h] += red[(tid + off) * NHEADS + h];
    __syncthreads();
  }
  float rcp[NHEADS];
#pragma unroll
  for (int h = 0; h < NHEADS; ++h)
    rcp[h] = 1.0f / fmaxf(red[h], 1e-30f);
  __syncthreads();

  // phase 3: normalize in place -> weights
  for (int i = s0 + tid; i < s1; i += 256)
#pragma unroll
    for (int h = 0; h < NHEADS; ++h) {
      const size_t idx = (size_t)i * NHEADS + h;
      scores[idx] = __expf(scores[idx] - smax[h]) * rcp[h];
    }
}

// ---------------------------------------------------------------- weighted pooling
__global__ __launch_bounds__(256)
void pool_kernel(const float* __restrict__ x, const float* __restrict__ w,
                 const int* __restrict__ seg, __bf16* __restrict__ pooled) {
  const int s = blockIdx.x;
  const int s0 = seg[s], s1 = seg[s + 1];
  const int tid = threadIdx.x;
  const int d0 = tid * 2;  // 512 dims / 256 threads
  __shared__ float wsm[64 * NHEADS];
  float acc[NHEADS][2];
#pragma unroll
  for (int h = 0; h < NHEADS; ++h) { acc[h][0] = 0.f; acc[h][1] = 0.f; }

  for (int base = s0; base < s1; base += 64) {
    const int cnt = (s1 - base < 64) ? (s1 - base) : 64;
    __syncthreads();
    if (tid < cnt * NHEADS)
      wsm[tid] = w[(size_t)(base + (tid >> 2)) * NHEADS + (tid & 3)];
    __syncthreads();
    for (int i = 0; i < cnt; ++i) {
      const float2 xv = *(const float2*)(x + (size_t)(base + i) * DIM + d0);
#pragma unroll
      for (int h = 0; h < NHEADS; ++h) {
        const float wv = wsm[i * NHEADS + h];
        acc[h][0] = fmaf(wv, xv.x, acc[h][0]);
        acc[h][1] = fmaf(wv, xv.y, acc[h][1]);
      }
    }
  }
#pragma unroll
  for (int h = 0; h < NHEADS; ++h) {
    pooled[(size_t)s * (NHEADS * DIM) + h * DIM + d0]     = (__bf16)acc[h][0];
    pooled[(size_t)s * (NHEADS * DIM) + h * DIM + d0 + 1] = (__bf16)acc[h][1];
  }
}

// ---------------------------------------------------------------- final GEMM (WMMA)
// out[1024,512] = pooled[1024,2048] @ Wc^T + bc ; one 16x16 tile per wave.
__global__ __launch_bounds__(256)
void out_gemm_kernel(const __bf16* __restrict__ A,    // [1024,2048]
                     const __bf16* __restrict__ Bw,   // Wc bf16 [512,2048]
                     const float* __restrict__ bc,    // [512]
                     float* __restrict__ out) {       // [1024,512]
  const int tid = threadIdx.x;
  const int lane = tid & 31;
  const int wave = tid >> 5;
  const int tile = blockIdx.x * 8 + wave;   // 2048 tiles total
  const int tM = tile >> 5;                 // 0..63
  const int tN = tile & 31;                 // 0..31
  const int mrow = tM * 16 + (lane & 15);
  const int ncol = tN * 16 + (lane & 15);
  const int kb    = (lane < 16) ? 0 : 16;
  const int bbase = (lane < 16) ? 0 : 16;

  v8f acc = {0.f, 0.f, 0.f, 0.f, 0.f, 0.f, 0.f, 0.f};
  for (int ks = 0; ks < 64; ++ks) {
    const int k0 = ks * 32;
    const v8bf alo = *(const v8bf*)(A + (size_t)mrow * 2048 + k0 + kb);
    const v8bf ahi = *(const v8bf*)(A + (size_t)mrow * 2048 + k0 + kb + 8);
    v16bf a;
#pragma unroll
    for (int i = 0; i < 8; ++i) { a[i] = alo[i]; a[8 + i] = ahi[i]; }
    const v16bf b = *(const v16bf*)(Bw + (size_t)ncol * 2048 + k0 + bbase);
    acc = wmma_bf16(a, b, acc);
  }
  const int rbase = (lane < 16) ? 0 : 8;
  const float bcv = bc[ncol];
#pragma unroll
  for (int r = 0; r < 8; ++r)
    out[(size_t)(tM * 16 + rbase + r) * DIM + ncol] = acc[r] + bcv;
}

// ---------------------------------------------------------------- launcher
extern "C" void kernel_launch(void* const* d_in, const int* in_sizes, int n_in,
                              void* d_out, int out_size, void* d_ws, size_t ws_size,
                              hipStream_t stream) {
  const float*     x     = (const float*)d_in[0];
  const long long* batch = (const long long*)d_in[1];
  const float*     Wh    = (const float*)d_in[2];
  const float*     bh    = (const float*)d_in[3];
  const float*     ctx   = (const float*)d_in[4];
  const float*     Wc    = (const float*)d_in[5];
  const float*     bc    = (const float*)d_in[6];
  float* out = (float*)d_out;

  char* ws = (char*)d_ws;
  __bf16* whb    = (__bf16*)(ws + WH_OFF);
  __bf16* wcb    = (__bf16*)(ws + WC_OFF);
  int*    seg    = (int*)(ws + SEG_OFF);
  float*  scores = (float*)(ws + SC_OFF);
  __bf16* pooled = (__bf16*)(ws + PL_OFF);

  const int nWh = NHEADS * HID * DIM;       // 131072
  const int nWc = DIM * NHEADS * DIM;       // 1048576

  cvt_f32_bf16_kernel<<<(nWh + 255) / 256, 256, 0, stream>>>(Wh, whb, nWh);
  cvt_f32_bf16_kernel<<<(nWc + 255) / 256, 256, 0, stream>>>(Wc, wcb, nWc);
  seg_bounds_kernel<<<(NSEG + 1 + 255) / 256, 256, 0, stream>>>(batch, seg);
  scores_kernel<<<NNODES / 128, 256, 0, stream>>>(x, whb, bh, ctx, scores);
  softmax_kernel<<<NSEG, 256, 0, stream>>>(scores, seg);
  pool_kernel<<<NSEG, 256, 0, stream>>>(x, scores, seg, pooled);
  out_gemm_kernel<<<(64 * 32) / 8, 256, 0, stream>>>(pooled, wcb, bc, out);
}
